// HGTImageFeatureExtractor_8916352106974
// MI455X (gfx1250) — matrix operand order repaired and checked by reading, not compiled, not run on hardware
//
#include <hip/hip_runtime.h>
#include <math.h>
#include <stdint.h>

// ---------------- problem constants ----------------
#define B_      256
#define P_      197
#define NPATCH  196
#define DIN     768
#define DD      256
#define HH      8
#define DHD     32
#define NVIT    50432      // B_*P_
#define NTEXT   4096
#define EVV     400000
#define EVT     100000
#define ETV     100000
#define ETOT    600000
#define NTOT    54528      // NVIT + NTEXT
#define FF_     1024
#define POOLK   98
#define INV_SQRT_DH 0.17677669529663687f

typedef __attribute__((ext_vector_type(16))) _Float16 v16h;
typedef __attribute__((ext_vector_type(8)))  float    v8f;

union AF { v16h v; _Float16 h[16]; };
union CF { v8f  v; float    f[8];  };

__device__ inline v8f wmma16(const v16h a, const v16h b, const v8f c) {
  return __builtin_amdgcn_wmma_f32_16x16x32_f16(false, a, false, b, (short)0, c,
                                                false, false);
}

__device__ inline float gelu_f(float x) {
  return 0.5f * x * (1.0f + erff(x * 0.70710678118654752f));
}

__device__ inline float wave_sum(float v) {
  #pragma unroll
  for (int m = 16; m; m >>= 1) v += __shfl_xor(v, m, 32);
  return v;
}

__device__ inline void atomicMaxFloat(float* addr, float value) {
  if (value >= 0.0f) atomicMax((int*)addr, __float_as_int(value));
  else               atomicMin((unsigned int*)addr, __float_as_uint(value));
}

// CDNA5 async DMA: global -> LDS, tracked by ASYNCcnt (ISA 08 §4).
// lds_off is a byte offset in LDS space (low 32 bits of a flat shared addr);
// g must be 8-byte aligned.
__device__ __forceinline__ void async_copy_b64(uint32_t lds_off, const float* g) {
  asm volatile("global_load_async_to_lds_b64 %0, %1, off"
               :: "v"(lds_off), "v"(g) : "memory");
}
__device__ __forceinline__ void wait_async0() {
  asm volatile("s_wait_asynccnt 0" ::: "memory");
}

// =====================================================================
// WMMA GEMM: C[M,N] = act( A[M,K] @ op(B) + bias[N] )
// transB==0: B is [K,N] row-major.  transB==1: B is [N,K] row-major (B^T).
// Block tile 128x64, 8 waves, each wave a 32x32 tile (4 x wmma 16x16x32 f16).
// Tiles staged fp32 in LDS by double-buffered GLOBAL_LOAD_ASYNC_TO_LDS_B64;
// f32->f16 conversion happens at fragment-load time.
// Requires: M%128==0, N%64==0, K%32==0 (true for every call here), and all
// global tile rows 8-byte aligned (K,N multiples of 32; base ptrs aligned).
// =====================================================================
__global__ __launch_bounds__(256) void gemm_wmma(
    const float* __restrict__ A, const float* __restrict__ Bm,
    const float* __restrict__ bias, float* __restrict__ C,
    int M, int N, int K, int transB, int act)
{
  // fp32 staging, double buffered. sA: [row][k] pitch 34.
  // sB: transB=1 -> [n][k] pitch 34 ; transB=0 -> [k][n] pitch 66 (2112<=2176).
  __shared__ float sA[2][128 * 34];
  __shared__ float sB[2][64 * 34];

  const int tid  = threadIdx.x;
  const int wave = tid >> 5;
  const int lane = tid & 31;
  const int m0 = blockIdx.y * 128;
  const int n0 = blockIdx.x * 64;
  const int wm = wave >> 1;          // 0..3 -> 32-row slab
  const int wn = wave & 1;           // 0..1 -> 32-col slab
  const int rl = lane & 15;
  const int hi = lane >> 4;

  const uint32_t ldsA[2] = { (uint32_t)(uintptr_t)&sA[0][0],
                             (uint32_t)(uintptr_t)&sA[1][0] };
  const uint32_t ldsB[2] = { (uint32_t)(uintptr_t)&sB[0][0],
                             (uint32_t)(uintptr_t)&sB[1][0] };

  auto issue = [&](int kc, int buf) {
    const uint32_t aB = ldsA[buf];
    const uint32_t bB = ldsB[buf];
    // A tile: 128 rows x 32 floats = 2048 x 8B chunks, 8 issues/thread
    #pragma unroll
    for (int i = 0; i < 8; i++) {
      int c = i * 256 + tid;
      int row = c >> 4, pr = (c & 15) * 2;
      const float* g = A + (size_t)(m0 + row) * K + kc + pr;
      async_copy_b64(aB + (uint32_t)(row * 34 + pr) * 4u, g);
    }
    if (transB) {   // B[N,K]: 64 rows x 32 floats
      #pragma unroll
      for (int i = 0; i < 4; i++) {
        int c = i * 256 + tid;
        int n = c >> 4, pr = (c & 15) * 2;
        const float* g = Bm + (size_t)(n0 + n) * K + kc + pr;
        async_copy_b64(bB + (uint32_t)(n * 34 + pr) * 4u, g);
      }
    } else {        // B[K,N]: 32 rows x 64 floats
      #pragma unroll
      for (int i = 0; i < 4; i++) {
        int c = i * 256 + tid;
        int k = c >> 5, pr = (c & 31) * 2;
        const float* g = Bm + (size_t)(kc + k) * N + n0 + pr;
        async_copy_b64(bB + (uint32_t)(k * 66 + pr) * 4u, g);
      }
    }
  };

  CF c[2][2];
  #pragma unroll
  for (int i = 0; i < 2; i++)
    #pragma unroll
    for (int j = 0; j < 2; j++)
      #pragma unroll
      for (int r = 0; r < 8; r++) c[i][j].f[r] = 0.0f;

  int cur = 0;
  issue(0, 0);
  for (int kc = 0; kc < K; kc += 32) {
    wait_async0();        // my async stores into sX[cur] complete
    __syncthreads();      // everyone's complete -> tile[cur] published
    if (kc + 32 < K) issue(kc + 32, cur ^ 1);   // overlap next tile DMA

    const float* aT = &sA[cur][0];
    const float* bT = &sB[cur][0];
    AF a0, a1, b0, b1;
    const int r0 = (wm * 32 + rl) * 34;
    const int r1 = (wm * 32 + 16 + rl) * 34;
    const int c0 = wn * 32 + rl;
    const int c1 = wn * 32 + 16 + rl;
    #pragma unroll
    for (int j = 0; j < 8; j++) {
      // 16-bit A fragment layout (ISA 7.12.2): lanes 0-15 K={2j | 16+2(j-4)},
      // lanes 16-31 +8.
      int ka = (j < 4 ? 2 * j : 8 + 2 * j) + hi * 8;
      a0.h[2*j]   = (_Float16)aT[r0 + ka];
      a0.h[2*j+1] = (_Float16)aT[r0 + ka + 1];
      a1.h[2*j]   = (_Float16)aT[r1 + ka];
      a1.h[2*j+1] = (_Float16)aT[r1 + ka + 1];
      // B fragment: lane -> column, same K order as A (lanes 0-15 K={0..7,16..23}, lanes 16-31 +8)
      int kb2 = (j < 4 ? 2 * j : 8 + 2 * j) + hi * 8;
      if (transB) {
        b0.h[2*j]   = (_Float16)bT[c0 * 34 + kb2];
        b0.h[2*j+1] = (_Float16)bT[c0 * 34 + kb2 + 1];
        b1.h[2*j]   = (_Float16)bT[c1 * 34 + kb2];
        b1.h[2*j+1] = (_Float16)bT[c1 * 34 + kb2 + 1];
      } else {
        b0.h[2*j]   = (_Float16)bT[kb2 * 66 + c0];
        b0.h[2*j+1] = (_Float16)bT[(kb2 + 1) * 66 + c0];
        b1.h[2*j]   = (_Float16)bT[kb2 * 66 + c1];
        b1.h[2*j+1] = (_Float16)bT[(kb2 + 1) * 66 + c1];
      }
    }
    c[0][0].v = wmma16(a0.v, b0.v, c[0][0].v);
    c[0][1].v = wmma16(a0.v, b1.v, c[0][1].v);
    c[1][0].v = wmma16(a1.v, b0.v, c[1][0].v);
    c[1][1].v = wmma16(a1.v, b1.v, c[1][1].v);
    __syncthreads();      // tile[cur] free for reuse (next-next DMA)
    cur ^= 1;
  }

  // epilogue: C/D layout: VGPR r -> row r + 8*hi, col = lane&15
  #pragma unroll
  for (int i = 0; i < 2; i++) {
    #pragma unroll
    for (int j = 0; j < 2; j++) {
      int colg = n0 + wn * 32 + j * 16 + rl;
      float bv = bias ? bias[colg] : 0.0f;
      int row0 = m0 + wm * 32 + i * 16 + hi * 8;
      #pragma unroll
      for (int r = 0; r < 8; r++) {
        float val = c[i][j].f[r] + bv;
        if (act == 1) val = gelu_f(val);
        C[(size_t)(row0 + r) * N + colg] = val;
      }
    }
  }
}

// =====================================================================
// Multi-head attention, one block per (batch, head). Online softmax,
// one query per thread, K/V staged in LDS (fp32).
// qkv rows have length 768: [q | k | v].
// =====================================================================
__global__ __launch_bounds__(256) void attn_kernel(
    const float* __restrict__ qkv, float* __restrict__ o)
{
  const int b = blockIdx.x;
  const int h = blockIdx.y;
  __shared__ float sK[P_ * DHD];
  __shared__ float sV[P_ * DHD];

  for (int idx = threadIdx.x; idx < P_ * DHD; idx += 256) {
    int p = idx >> 5, d = idx & 31;
    const float* row = qkv + (size_t)(b * P_ + p) * 768;
    sK[idx] = row[256 + h * DHD + d];
    sV[idx] = row[512 + h * DHD + d];
  }
  __syncthreads();

  int q = threadIdx.x;
  if (q >= P_) return;

  float qv[DHD];
  const float* qr = qkv + (size_t)(b * P_ + q) * 768 + h * DHD;
  #pragma unroll
  for (int d = 0; d < DHD; d++) qv[d] = qr[d];

  float m = -INFINITY, l = 0.0f, acc[DHD];
  #pragma unroll
  for (int d = 0; d < DHD; d++) acc[d] = 0.0f;

  for (int k = 0; k < P_; k++) {
    float s = 0.0f;
    #pragma unroll
    for (int d = 0; d < DHD; d++) s += qv[d] * sK[k * DHD + d];
    s *= INV_SQRT_DH;
    float mn = fmaxf(m, s);
    float corr = expf(m - mn);      // exp(-inf)=0 on first iter
    float e = expf(s - mn);
    l = l * corr + e;
    #pragma unroll
    for (int d = 0; d < DHD; d++) acc[d] = acc[d] * corr + e * sV[k * DHD + d];
    m = mn;
  }
  float inv = 1.0f / l;
  float* orow = o + (size_t)(b * P_ + q) * DD + h * DHD;
  #pragma unroll
  for (int d = 0; d < DHD; d++) orow[d] = acc[d] * inv;
}

// ---------------- x = LayerNorm(x + o) ; one wave per 256-wide row ----
__global__ __launch_bounds__(256) void ln_residual_k(
    float* __restrict__ x, const float* __restrict__ o,
    const float* __restrict__ g, const float* __restrict__ bb, int nrows)
{
  int wave = threadIdx.x >> 5, lane = threadIdx.x & 31;
  int row = blockIdx.x * 8 + wave;
  if (row >= nrows) return;
  size_t base = (size_t)row * DD;
  float v[8], s = 0.0f;
  #pragma unroll
  for (int j = 0; j < 8; j++) {
    int cc = lane + 32 * j;
    v[j] = x[base + cc] + o[base + cc];
    s += v[j];
  }
  float mu = wave_sum(s) * (1.0f / 256.0f);
  float var = 0.0f;
  #pragma unroll
  for (int j = 0; j < 8; j++) { float d = v[j] - mu; var += d * d; }
  var = wave_sum(var) * (1.0f / 256.0f);
  float rs = rsqrtf(var + 1e-5f);
  #pragma unroll
  for (int j = 0; j < 8; j++) {
    int cc = lane + 32 * j;
    x[base + cc] = (v[j] - mu) * rs * g[cc] + bb[cc];
  }
}

// ---- HGT update: h = h + relu(LN(sk*o + (1-sk)*h)) ------------------
__global__ __launch_bounds__(256) void hgt_update_k(
    float* __restrict__ hb, const float* __restrict__ o,
    const float* __restrict__ g, const float* __restrict__ bb,
    const float* __restrict__ skipp, int nrows)
{
  int wave = threadIdx.x >> 5, lane = threadIdx.x & 31;
  int row = blockIdx.x * 8 + wave;
  if (row >= nrows) return;
  float sk = 1.0f / (1.0f + expf(-skipp[0]));
  size_t base = (size_t)row * DD;
  float hv[8], u[8], s = 0.0f;
  #pragma unroll
  for (int j = 0; j < 8; j++) {
    int cc = lane + 32 * j;
    hv[j] = hb[base + cc];
    u[j] = sk * o[base + cc] + (1.0f - sk) * hv[j];
    s += u[j];
  }
  float mu = wave_sum(s) * (1.0f / 256.0f);
  float var = 0.0f;
  #pragma unroll
  for (int j = 0; j < 8; j++) { float d = u[j] - mu; var += d * d; }
  var = wave_sum(var) * (1.0f / 256.0f);
  float rs = rsqrtf(var + 1e-5f);
  #pragma unroll
  for (int j = 0; j < 8; j++) {
    int cc = lane + 32 * j;
    float t = (u[j] - mu) * rs * g[cc] + bb[cc];
    hb[base + cc] = hv[j] + fmaxf(t, 0.0f);
  }
}

// ---------------- elementwise exact GELU -----------------------------
__global__ void gelu_k(const float* __restrict__ in, float* __restrict__ out,
                       size_t n)
{
  size_t i = (size_t)blockIdx.x * 256 + threadIdx.x;
  if (i < n) out[i] = gelu_f(in[i]);
}

// ---------------- segment id precompute ------------------------------
__global__ void seg_k(const int* __restrict__ vv, const int* __restrict__ vt,
                      const int* __restrict__ tv, int* __restrict__ seg)
{
  int e = blockIdx.x * 256 + threadIdx.x;
  if (e >= ETOT) return;
  int s;
  if (e < EVV)            s = vv[EVV + e];
  else if (e < EVV + EVT) s = vt[EVT + (e - EVV)] + NVIT;
  else                    s = tv[ETV + (e - EVV - EVT)];
  seg[e] = s;
}

// ---------------- per-layer init: agg=0, m=-inf, den=0 ---------------
__global__ void hgt_init_k(float* __restrict__ agg, float* __restrict__ m,
                           float* __restrict__ den)
{
  size_t tid = (size_t)blockIdx.x * 256 + threadIdx.x;
  if (tid < (size_t)NTOT * DD) agg[tid] = 0.0f;
  if (tid < (size_t)NTOT * HH) { m[tid] = -INFINITY; den[tid] = 0.0f; }
}

// ---- edge logits: wave = head, block handles 8 edges ----------------
// logit = p_rel * invsqrt * dot(q[dst], A^T k[src])
__global__ __launch_bounds__(256) void hgt_edge_logits_k(
    const int* __restrict__ ei, int E, int goff,
    const float* __restrict__ ksrc, const float* __restrict__ qdst,
    const float* __restrict__ Arel, const float* __restrict__ prel,
    float* __restrict__ lg)
{
  int h = threadIdx.x >> 5, lane = threadIdx.x & 31;
  const float* Ah = Arel + h * 1024;
  float Acol[32];
  #pragma unroll
  for (int d = 0; d < 32; d++) Acol[d] = Ah[d * 32 + lane];
  float pr = prel[h] * INV_SQRT_DH;
  int base = blockIdx.x * 8;
  for (int i = 0; i < 8; i++) {
    int e = base + i;
    if (e >= E) break;
    int s  = ei[e];
    int dd = ei[E + e];
    float kv = ksrc[(size_t)s * 768 + h * DHD + lane];   // k block = cols [0,256)
    float acc = 0.0f;
    #pragma unroll
    for (int d = 0; d < 32; d++) acc += __shfl(kv, d, 32) * Acol[d];
    float qf = qdst[(size_t)dd * 768 + 256 + h * DHD + lane]; // q block
    float p = wave_sum(qf * acc);
    if (lane == 0) lg[(size_t)(goff + e) * HH + h] = p * pr;
  }
}

// ---- segment max (atomic) ------------------------------------------
__global__ void hgt_amax_k(const float* __restrict__ lg,
                           const int* __restrict__ seg, float* __restrict__ m)
{
  int tid = blockIdx.x * 256 + threadIdx.x;
  if (tid >= ETOT * HH) return;
  int e = tid >> 3, h = tid & 7;
  atomicMaxFloat(&m[seg[e] * HH + h], lg[tid]);
}

// ---- ex = exp(lg - m[seg]); den += ex (atomic); lg <- ex ------------
__global__ void hgt_exden_k(float* __restrict__ lg, const int* __restrict__ seg,
                            const float* __restrict__ m, float* __restrict__ den)
{
  int tid = blockIdx.x * 256 + threadIdx.x;
  if (tid >= ETOT * HH) return;
  int e = tid >> 3, h = tid & 7;
  int sg = seg[e];
  float ex = expf(lg[tid] - m[sg * HH + h]);
  lg[tid] = ex;
  atomicAdd(&den[sg * HH + h], ex);
}

// ---- accumulate: agg[seg] += alpha * (M^T v[src]) (v recomputed) ----
__global__ __launch_bounds__(256) void hgt_accum_k(
    const int* __restrict__ ei, int E, int goff,
    const float* __restrict__ vsrc, const float* __restrict__ Mrel,
    const int* __restrict__ seg, const float* __restrict__ den,
    const float* __restrict__ lgex, float* __restrict__ agg)
{
  int h = threadIdx.x >> 5, lane = threadIdx.x & 31;
  const float* Mh = Mrel + h * 1024;
  float Mcol[32];
  #pragma unroll
  for (int d = 0; d < 32; d++) Mcol[d] = Mh[d * 32 + lane];
  int base = blockIdx.x * 8;
  for (int i = 0; i < 8; i++) {
    int e = base + i;
    if (e >= E) break;
    int s = ei[e];
    float vv = vsrc[(size_t)s * 768 + 512 + h * DHD + lane]; // v block
    float acc = 0.0f;
    #pragma unroll
    for (int d = 0; d < 32; d++) acc += __shfl(vv, d, 32) * Mcol[d];
    int sg = seg[goff + e];
    float ex = lgex[(size_t)(goff + e) * HH + h];
    float dn = den[sg * HH + h];
    atomicAdd(&agg[(size_t)sg * DD + h * DHD + lane], acc * (ex / (dn + 1e-16f)));
  }
}

// ---------------- saliency scores (wave per patch row) ---------------
__global__ __launch_bounds__(256) void scores_k(
    const float* __restrict__ hvit, const float* __restrict__ salw,
    const float* __restrict__ salb, float* __restrict__ scores)
{
  int wave = threadIdx.x >> 5, lane = threadIdx.x & 31;
  int row = blockIdx.x * 8 + wave;
  if (row >= B_ * NPATCH) return;
  int b = row / NPATCH, p = row % NPATCH;
  const float* vr = hvit + (size_t)(b * P_ + 1 + p) * DD;
  float s = 0.0f;
  #pragma unroll
  for (int j = 0; j < 8; j++) { int cc = lane + 32 * j; s += vr[cc] * salw[cc]; }
  s = wave_sum(s);
  if (lane == 0) scores[row] = s + salb[0];
}

// ---------------- top-98 pooling, one block per batch ----------------
__global__ __launch_bounds__(256) void pool_k(
    const float* __restrict__ hvit, const float* __restrict__ scores,
    float* __restrict__ out)
{
  int b = blockIdx.x;
  __shared__ float sc[NPATCH];
  __shared__ int sel[NPATCH];
  int t = threadIdx.x;
  if (t < NPATCH) sc[t] = scores[b * NPATCH + t];
  __syncthreads();
  if (t < NPATCH) {
    float si = sc[t];
    int rank = 0;
    for (int j = 0; j < NPATCH; j++) {
      float sj = sc[j];
      rank += (sj > si) || (sj == si && j < t);   // stable-descending rank
    }
    sel[t] = (rank < POOLK) ? 1 : 0;
  }
  __syncthreads();
  float acc = 0.0f;
  for (int p = 0; p < NPATCH; p++)
    if (sel[p]) acc += hvit[(size_t)(b * P_ + 1 + p) * DD + t];
  out[(size_t)b * DD + t] = hvit[(size_t)(b * P_) * DD + t] + acc * (1.0f / POOLK);
}

// =====================================================================
// host side
// =====================================================================
static inline void gemm(const float* A, const float* Bm, const float* bias,
                        float* C, int M, int N, int K, int transB, int act,
                        hipStream_t s)
{
  dim3 g(N / 64, M / 128);
  gemm_wmma<<<g, 256, 0, s>>>(A, Bm, bias, C, M, N, K, transB, act);
}

extern "C" void kernel_launch(void* const* d_in, const int* in_sizes, int n_in,
                              void* d_out, int out_size, void* d_ws, size_t ws_size,
                              hipStream_t stream)
{
  (void)in_sizes; (void)n_in; (void)out_size; (void)ws_size;
  const float* x_vit        = (const float*)d_in[0];
  const float* x_text       = (const float*)d_in[1];
  const int*   ei_vv        = (const int*)  d_in[2];
  const int*   ei_vt        = (const int*)  d_in[3];
  const int*   ei_tv        = (const int*)  d_in[4];
  const float* proj_w_vit   = (const float*)d_in[6];
  const float* proj_b_vit   = (const float*)d_in[7];
  const float* proj_w_text  = (const float*)d_in[8];
  const float* proj_b_text  = (const float*)d_in[9];
  const float* tr_qkv_w     = (const float*)d_in[10];
  const float* tr_qkv_b     = (const float*)d_in[11];
  const float* tr_out_w     = (const float*)d_in[12];
  const float* tr_out_b     = (const float*)d_in[13];
  const float* tr_ln1_g     = (const float*)d_in[14];
  const float* tr_ln1_b     = (const float*)d_in[15];
  const float* tr_ff1_w     = (const float*)d_in[16];
  const float* tr_ff1_b     = (const float*)d_in[17];
  const float* tr_ff2_w     = (const float*)d_in[18];
  const float* tr_ff2_b     = (const float*)d_in[19];
  const float* tr_ln2_g     = (const float*)d_in[20];
  const float* tr_ln2_b     = (const float*)d_in[21];
  const float* hgt_kqv_w_v  = (const float*)d_in[22];
  const float* hgt_kqv_b_v  = (const float*)d_in[23];
  const float* hgt_kqv_w_t  = (const float*)d_in[24];
  const float* hgt_kqv_b_t  = (const float*)d_in[25];
  const float* hgt_a_rel    = (const float*)d_in[26];
  const float* hgt_m_rel    = (const float*)d_in[27];
  const float* hgt_p_rel    = (const float*)d_in[28];
  const float* hgt_out_w_v  = (const float*)d_in[29];
  const float* hgt_out_b_v  = (const float*)d_in[30];
  const float* hgt_out_w_t  = (const float*)d_in[31];
  const float* hgt_out_b_t  = (const float*)d_in[32];
  const float* hgt_skip_v   = (const float*)d_in[33];
  const float* hgt_skip_t   = (const float*)d_in[34];
  const float* ln_g_vit     = (const float*)d_in[35];
  const float* ln_b_vit     = (const float*)d_in[36];
  const float* ln_g_text    = (const float*)d_in[37];
  const float* ln_b_text    = (const float*)d_in[38];
  const float* sal_w        = (const float*)d_in[39];
  const float* sal_b        = (const float*)d_in[40];

  // -------- workspace layout (floats), total ~128M floats (~511 MB) --
  float* ws = (float*)d_ws;
  size_t off = 0;
  auto alloc = [&](size_t n) { float* p = ws + off; off += n; return p; };
  float* h_vit = alloc((size_t)NVIT * DD);
  float* h_text = alloc((size_t)NTEXT * DD);
  float* qkv   = alloc((size_t)NVIT * 768);      // TR qkv / HGT kqv_v
  float* tmp1  = alloc((size_t)NVIT * FF_);      // FF inter / HGT agg / out_v
  float* tmp2  = alloc((size_t)NVIT * DD);       // attn out / gelu(agg_v)
  float* kqvt  = alloc((size_t)NTEXT * 768);     // HGT kqv_t / gelu(agg_t)
  float* lg    = alloc((size_t)ETOT * HH);       // logits, then ex
  float* mbuf  = alloc((size_t)NTOT * HH);
  float* denb  = alloc((size_t)NTOT * HH);
  float* outt  = alloc((size_t)NTEXT * DD);      // HGT out_t
  float* scrs  = alloc((size_t)B_ * NPATCH);
  int*   segb  = (int*)alloc((size_t)ETOT);

  // -------- segment ids (layer-invariant) ---------------------------
  seg_k<<<(ETOT + 255) / 256, 256, 0, stream>>>(ei_vv, ei_vt, ei_tv, segb);

  // -------- input projections ---------------------------------------
  gemm(x_vit,  proj_w_vit,  proj_b_vit,  h_vit,  NVIT,  DD, DIN, 0, 0, stream);
  gemm(x_text, proj_w_text, proj_b_text, h_text, NTEXT, DD, DIN, 0, 0, stream);

  // -------- transformer layers on h_vit ------------------------------
  for (int l = 0; l < 2; l++) {
    gemm(h_vit, tr_qkv_w + (size_t)l * 768 * 256, tr_qkv_b + l * 768,
         qkv, NVIT, 768, 256, 1, 0, stream);
    attn_kernel<<<dim3(B_, HH), 256, 0, stream>>>(qkv, tmp2);
    gemm(tmp2, tr_out_w + (size_t)l * 256 * 256, tr_out_b + l * 256,
         tmp1, NVIT, 256, 256, 1, 0, stream);
    ln_residual_k<<<NVIT / 8, 256, 0, stream>>>(h_vit, tmp1,
        tr_ln1_g + l * 256, tr_ln1_b + l * 256, NVIT);
    gemm(h_vit, tr_ff1_w + (size_t)l * 1024 * 256, tr_ff1_b + l * 1024,
         tmp1, NVIT, 1024, 256, 1, 1, stream);          // fused GELU
    gemm(tmp1, tr_ff2_w + (size_t)l * 256 * 1024, tr_ff2_b + l * 256,
         tmp2, NVIT, 256, 1024, 1, 0, stream);
    ln_residual_k<<<NVIT / 8, 256, 0, stream>>>(h_vit, tmp2,
        tr_ln2_g + l * 256, tr_ln2_b + l * 256, NVIT);
  }

  // -------- HGT layers ----------------------------------------------
  float* agg = tmp1;                // NTOT x 256 fits inside tmp1
  for (int l = 0; l < 2; l++) {
    gemm(h_vit,  hgt_kqv_w_v + (size_t)l * 256 * 768, hgt_kqv_b_v + l * 768,
         qkv,  NVIT,  768, 256, 0, 0, stream);
    gemm(h_text, hgt_kqv_w_t + (size_t)l * 256 * 768, hgt_kqv_b_t + l * 768,
         kqvt, NTEXT, 768, 256, 0, 0, stream);

    hgt_init_k<<<NTOT, 256, 0, stream>>>(agg, mbuf, denb);

    const float* Ab = hgt_a_rel + (size_t)(l * 3) * 8192;
    const float* Mb = hgt_m_rel + (size_t)(l * 3) * 8192;
    const float* Pb = hgt_p_rel + (size_t)(l * 3) * 8;
    // e=0: vit->vit, e=1: vit->text, e=2: text->vit
    hgt_edge_logits_k<<<EVV / 8, 256, 0, stream>>>(ei_vv, EVV, 0,
        qkv,  qkv,  Ab + 0 * 8192, Pb + 0 * 8, lg);
    hgt_edge_logits_k<<<EVT / 8, 256, 0, stream>>>(ei_vt, EVT, EVV,
        qkv,  kqvt, Ab + 1 * 8192, Pb + 1 * 8, lg);
    hgt_edge_logits_k<<<ETV / 8, 256, 0, stream>>>(ei_tv, ETV, EVV + EVT,
        kqvt, qkv,  Ab + 2 * 8192, Pb + 2 * 8, lg);

    hgt_amax_k <<<(ETOT * HH + 255) / 256, 256, 0, stream>>>(lg, segb, mbuf);
    hgt_exden_k<<<(ETOT * HH + 255) / 256, 256, 0, stream>>>(lg, segb, mbuf, denb);

    hgt_accum_k<<<EVV / 8, 256, 0, stream>>>(ei_vv, EVV, 0,
        qkv,  Mb + 0 * 8192, segb, denb, lg, agg);
    hgt_accum_k<<<EVT / 8, 256, 0, stream>>>(ei_vt, EVT, EVV,
        qkv,  Mb + 1 * 8192, segb, denb, lg, agg);
    hgt_accum_k<<<ETV / 8, 256, 0, stream>>>(ei_tv, ETV, EVV + EVT,
        kqvt, Mb + 2 * 8192, segb, denb, lg, agg);

    // gelu on aggregated messages, then output projections
    gelu_k<<<((size_t)NVIT * DD + 255) / 256, 256, 0, stream>>>(
        agg, tmp2, (size_t)NVIT * DD);
    gelu_k<<<((size_t)NTEXT * DD + 255) / 256, 256, 0, stream>>>(
        agg + (size_t)NVIT * DD, kqvt, (size_t)NTEXT * DD);  // kqvt reused
    gemm(tmp2, hgt_out_w_v + (size_t)l * 256 * 256, hgt_out_b_v + l * 256,
         tmp1, NVIT, 256, 256, 0, 0, stream);                // tmp1 = out_v
    gemm(kqvt, hgt_out_w_t + (size_t)l * 256 * 256, hgt_out_b_t + l * 256,
         outt, NTEXT, 256, 256, 0, 0, stream);

    hgt_update_k<<<NVIT / 8, 256, 0, stream>>>(h_vit, tmp1,
        ln_g_vit + l * 256, ln_b_vit + l * 256, hgt_skip_v + l, NVIT);
    hgt_update_k<<<NTEXT / 8, 256, 0, stream>>>(h_text, outt,
        ln_g_text + l * 256, ln_b_text + l * 256, hgt_skip_t + l, NTEXT);
  }

  // -------- saliency + top-k pooling ---------------------------------
  scores_k<<<(B_ * NPATCH) / 8, 256, 0, stream>>>(h_vit, sal_w, sal_b, scrs);
  pool_k<<<B_, 256, 0, stream>>>(h_vit, scrs, (float*)d_out);

  // -------- h_text -> second output ----------------------------------
  hipMemcpyAsync((float*)d_out + (size_t)B_ * DD, h_text,
                 sizeof(float) * (size_t)NTEXT * DD,
                 hipMemcpyDeviceToDevice, stream);
}
